// IsoMaxLossFirstPart_59081570124167
// MI455X (gfx1250) — matrix operand order repaired and checked by reading, not compiled, not run on hardware
//
#include <hip/hip_runtime.h>
#include <math.h>

// ---------------------------------------------------------------------------
// IsoMax first part:  logits[b][c] = -sqrt(max(|f_b|^2 + |p_c|^2 - 2 f_b.p_c, 0))
// B=16384, F=2048, C=1000, fp32.
// Split-bf16 WMMA GEMM (hi*hi + lo*hi + hi*lo), norms precomputed in d_ws.
// ---------------------------------------------------------------------------

typedef __attribute__((ext_vector_type(16))) __bf16 v16bf;
typedef __attribute__((ext_vector_type(8)))  float  v8f;

#define FDIM   2048
#define BM     128
#define BN     128
#define BK     32
#define KSTEPS (FDIM / BK)   // 64
#define ROWW   20            // u32 words per LDS tile row: 16 data + 4 pad (80B)

struct alignas(16) F4  { float x, y, z, w; };
struct alignas(16) U4  { unsigned int x, y, z, w; };
struct alignas(16) U8  { U4 a, b; };          // 32 bytes == one v16bf fragment

// ---- fp32 -> bf16 (RNE) helpers -------------------------------------------
__device__ __forceinline__ unsigned int bf16_rne(float x) {
  unsigned int u = __builtin_bit_cast(unsigned int, x);
  unsigned int r = u + 0x7FFFu + ((u >> 16) & 1u);
  return (r >> 16) & 0xFFFFu;
}
__device__ __forceinline__ float bf16_as_f32(unsigned int h16) {
  return __builtin_bit_cast(float, h16 << 16);
}
// pack two consecutive-K floats into one hi word and one lo word
__device__ __forceinline__ void cvt_pair(float x0, float x1,
                                         unsigned int& hw, unsigned int& lw) {
  unsigned int h0 = bf16_rne(x0);
  unsigned int h1 = bf16_rne(x1);
  float r0 = x0 - bf16_as_f32(h0);
  float r1 = x1 - bf16_as_f32(h1);
  unsigned int l0 = bf16_rne(r0);
  unsigned int l1 = bf16_rne(r1);
  hw = h0 | (h1 << 16);
  lw = l0 | (l1 << 16);
}

// ---- fragment loads from LDS ----------------------------------------------
// 16-bit A matrix 16x32 (ISA 7.12.2): lane half 0 holds K[0..7] and K[16..23],
// lane half 1 holds K[8..15] and K[24..31]  (chunks of 8 bf16 = 16B = 4 words)
__device__ __forceinline__ v16bf ldfragA(const unsigned int* rowp, int khalf) {
  U8 u;
  u.a = *(const U4*)(rowp + khalf * 4);
  u.b = *(const U4*)(rowp + 8 + khalf * 4);
  return __builtin_bit_cast(v16bf, u);
}
// 16-bit B matrix 32x16: lane half 0 holds K[0..7] and K[16..23], half 1 holds K[8..15] and K[24..31]
__device__ __forceinline__ v16bf ldfragB(const unsigned int* rowp, int khalf) {
  U8 u;
  u.a = *(const U4*)(rowp + khalf * 4);
  u.b = *(const U4*)(rowp + 8 + khalf * 4);
  return __builtin_bit_cast(v16bf, u);
}

// ---------------------------------------------------------------------------
// Row-norm kernel: one wave (32 lanes) per row of a [nrows x 2048] matrix.
// ---------------------------------------------------------------------------
__global__ void __launch_bounds__(256)
rownorm_kernel(const float* __restrict__ src, float* __restrict__ out, int nrows) {
  const int w    = threadIdx.x >> 5;
  const int lane = threadIdx.x & 31;
  const int row  = blockIdx.x * 8 + w;
  if (row >= nrows) return;
  const F4* p = (const F4*)(src + (size_t)row * FDIM);
  float s = 0.0f;
  #pragma unroll 4
  for (int i = lane; i < FDIM / 4; i += 32) {
    F4 v = p[i];
    s += v.x * v.x + v.y * v.y + v.z * v.z + v.w * v.w;
  }
  #pragma unroll
  for (int m = 16; m >= 1; m >>= 1) s += __shfl_xor(s, m, 32);
  if (lane == 0) out[row] = s;
}

// ---------------------------------------------------------------------------
// Main split-bf16 WMMA GEMM + distance epilogue.
// Block tile 128x128, 8 waves in a 4(M) x 2(N) grid; wave tile 32x64.
// Pad rows of B are index-clamped (not zeroed): they only contaminate output
// columns that the epilogue never stores.
// ---------------------------------------------------------------------------
__global__ void __launch_bounds__(256)
isomax_gemm_kernel(const float* __restrict__ feat, const float* __restrict__ proto,
                   const float* __restrict__ fsq, const float* __restrict__ psq,
                   float* __restrict__ out, int Bsz, int Csz) {
  __shared__ unsigned int sAh[BM * ROWW];
  __shared__ unsigned int sAl[BM * ROWW];
  __shared__ unsigned int sBh[BN * ROWW];
  __shared__ unsigned int sBl[BN * ROWW];
  __shared__ float sFs[BM];
  __shared__ float sPs[BN];

  const int tid = threadIdx.x;
  const int m0  = blockIdx.y * BM;
  const int n0  = blockIdx.x * BN;

  // stage squared norms for this block (guarded)
  if (tid < BM) {
    int m = m0 + tid;
    sFs[tid] = (m < Bsz) ? fsq[m] : 0.0f;
  } else {
    int c = tid - BM;
    int n = n0 + c;
    sPs[c] = (n < Csz) ? psq[n] : 0.0f;
  }

  // --- global staging: 2 threads per tile row, 16 floats each; clamp rows ---
  const int row2  = tid >> 1;          // 0..127
  const int kseg  = (tid & 1) * 16;    // 0 or 16 (element offset within BK)
  const int aRowG = min(m0 + row2, Bsz - 1);
  const int bRowG = min(n0 + row2, Csz - 1);
  const F4* aPtr  = (const F4*)(feat  + (size_t)aRowG * FDIM + kseg);
  const F4* bPtr  = (const F4*)(proto + (size_t)bRowG * FDIM + kseg);
  const int wbase = row2 * ROWW + (kseg >> 1);   // u32 word base in LDS row

  // --- wave / lane coordinates ---
  const int w     = tid >> 5;
  const int lane  = tid & 31;
  const int waveM = w & 3;    // 0..3 -> 32-row slabs
  const int waveN = w >> 2;   // 0..1 -> 64-col slabs
  const int lrow  = lane & 15;
  const int khalf = lane >> 4;

  const v8f vzero = {0.f, 0.f, 0.f, 0.f, 0.f, 0.f, 0.f, 0.f};
  v8f acc[2][4];
  #pragma unroll
  for (int mf = 0; mf < 2; ++mf)
    #pragma unroll
    for (int nf = 0; nf < 4; ++nf) acc[mf][nf] = vzero;

  for (int kt = 0; kt < KSTEPS; ++kt) {
    // ---- load this K-tile (fp32) and convert immediately to hi/lo words ----
    unsigned int hA[8], lA[8], hB[8], lB[8];
    {
      F4 aV[4], bV[4];
      #pragma unroll
      for (int j = 0; j < 4; ++j) {
        aV[j] = aPtr[kt * 8 + j];
        bV[j] = bPtr[kt * 8 + j];
      }
      // speculative prefetch of the next K-tile (safe: dropped on fault)
      __builtin_prefetch(&aPtr[kt * 8 + 8], 0, 0);
      __builtin_prefetch(&bPtr[kt * 8 + 8], 0, 0);
      #pragma unroll
      for (int j = 0; j < 4; ++j) {
        cvt_pair(aV[j].x, aV[j].y, hA[j * 2 + 0], lA[j * 2 + 0]);
        cvt_pair(aV[j].z, aV[j].w, hA[j * 2 + 1], lA[j * 2 + 1]);
        cvt_pair(bV[j].x, bV[j].y, hB[j * 2 + 0], lB[j * 2 + 0]);
        cvt_pair(bV[j].z, bV[j].w, hB[j * 2 + 1], lB[j * 2 + 1]);
      }
    }

    __syncthreads();                 // previous iteration finished reading LDS

    *(U4*)&sAh[wbase]     = U4{hA[0], hA[1], hA[2], hA[3]};
    *(U4*)&sAh[wbase + 4] = U4{hA[4], hA[5], hA[6], hA[7]};
    *(U4*)&sAl[wbase]     = U4{lA[0], lA[1], lA[2], lA[3]};
    *(U4*)&sAl[wbase + 4] = U4{lA[4], lA[5], lA[6], lA[7]};
    *(U4*)&sBh[wbase]     = U4{hB[0], hB[1], hB[2], hB[3]};
    *(U4*)&sBh[wbase + 4] = U4{hB[4], hB[5], hB[6], hB[7]};
    *(U4*)&sBl[wbase]     = U4{lB[0], lB[1], lB[2], lB[3]};
    *(U4*)&sBl[wbase + 4] = U4{lB[4], lB[5], lB[6], lB[7]};

    __syncthreads();

    // ---- A fragments stay live; B fragments streamed one pair at a time ----
    v16bf ah[2], al[2];
    #pragma unroll
    for (int mf = 0; mf < 2; ++mf) {
      int r = waveM * 32 + mf * 16 + lrow;
      ah[mf] = ldfragA(&sAh[r * ROWW], khalf);
      al[mf] = ldfragA(&sAl[r * ROWW], khalf);
    }
    #pragma unroll
    for (int nf = 0; nf < 4; ++nf) {
      int r = waveN * 64 + nf * 16 + lrow;
      v16bf bh = ldfragB(&sBh[r * ROWW], khalf);
      v16bf bl = ldfragB(&sBl[r * ROWW], khalf);
      #pragma unroll
      for (int mf = 0; mf < 2; ++mf) {
        acc[mf][nf] = __builtin_amdgcn_wmma_f32_16x16x32_bf16(
            false, ah[mf], false, bh, (short)0, acc[mf][nf], false, false);
        acc[mf][nf] = __builtin_amdgcn_wmma_f32_16x16x32_bf16(
            false, al[mf], false, bh, (short)0, acc[mf][nf], false, false);
        acc[mf][nf] = __builtin_amdgcn_wmma_f32_16x16x32_bf16(
            false, ah[mf], false, bl, (short)0, acc[mf][nf], false, false);
      }
    }
  }

  // --- epilogue: d = -sqrt(max(|f|^2 + |p|^2 - 2*cross, 0)) ---
  // C/D layout: VGPR r holds row (khalf*8 + r), column = lane & 15.
  #pragma unroll
  for (int nf = 0; nf < 4; ++nf) {
    const int colLocal = waveN * 64 + nf * 16 + lrow;
    const int n = n0 + colLocal;
    if (n >= Csz) continue;
    const float pn = sPs[colLocal];
    #pragma unroll
    for (int mf = 0; mf < 2; ++mf) {
      #pragma unroll
      for (int r = 0; r < 8; ++r) {
        const int rowLocal = waveM * 32 + mf * 16 + khalf * 8 + r;
        const int m = m0 + rowLocal;
        if (m < Bsz) {
          float d2 = __builtin_fmaf(-2.0f, acc[mf][nf][r], sFs[rowLocal] + pn);
          d2 = fmaxf(d2, 0.0f);
          out[(size_t)m * Csz + n] = -__fsqrt_rn(d2);
        }
      }
    }
  }
}

// ---------------------------------------------------------------------------
extern "C" void kernel_launch(void* const* d_in, const int* in_sizes, int n_in,
                              void* d_out, int out_size, void* d_ws, size_t ws_size,
                              hipStream_t stream) {
  const float* features   = (const float*)d_in[0];
  const float* prototypes = (const float*)d_in[1];
  float*       out        = (float*)d_out;

  const int Bsz = in_sizes[0] / FDIM;   // 16384
  const int Csz = in_sizes[1] / FDIM;   // 1000

  float* fsq = (float*)d_ws;            // [Bsz]
  float* psq = fsq + Bsz;               // [Csz]

  rownorm_kernel<<<(Bsz + 7) / 8, 256, 0, stream>>>(features,   fsq, Bsz);
  rownorm_kernel<<<(Csz + 7) / 8, 256, 0, stream>>>(prototypes, psq, Csz);

  dim3 grid((Csz + BN - 1) / BN, (Bsz + BM - 1) / BM);   // (8, 128)
  isomax_gemm_kernel<<<grid, 256, 0, stream>>>(features, prototypes,
                                               fsq, psq, out, Bsz, Csz);
}
